// DPLSTMModel_86603720556998
// MI455X (gfx1250) — compile-verified, matrix operand order checked by reading, not run on hardware
//
#include <hip/hip_runtime.h>
#include <hip/hip_bf16.h>

#define VOCAB 50257
#define E     1024
#define H     1024
#define Bsz   16
#define Tlen  256
#define BT    (Bsz * Tlen)      /* 4096 */
#define G4H   4096
#define NPAD  50304             /* 3144 N-tiles of 16 for decoder */
#define NGRP  (NPAD / 64)       /* 786 groups of 4 N-tiles */
#define LOGP_N ((long long)BT * VOCAB)

typedef __bf16 bf16_t;
typedef __attribute__((ext_vector_type(16))) __bf16 v16bf;
typedef __attribute__((ext_vector_type(8)))  float  v8f;

struct Frag32 { uint4 lo, hi; };   // 32 bytes = 16 bf16 = one WMMA A/B fragment per lane

static __device__ __forceinline__ v16bf frag_from(const unsigned short* p) {
  Frag32 f;
  f.lo = *(const uint4*)(p);
  f.hi = *(const uint4*)(p + 16);
  return __builtin_bit_cast(v16bf, f);
}

// One 16-bit WMMA A/B fragment (ISA 7.12.2 layout): lane L holds row/col (L&15);
// lanes 0-15 take K = [k0..k0+7] ++ [k0+16..k0+23], lanes 16-31 take [k0+8..k0+15] ++ [k0+24..k0+31].
static __device__ __forceinline__ v16bf load_frag(const unsigned short* __restrict__ base,
                                                  int stride, int k0, int lane) {
  return frag_from(base + (size_t)(lane & 15) * stride + k0 + ((lane >> 4) << 3));
}

static __device__ __forceinline__ unsigned short f2bf(float x) {
  unsigned int u = __float_as_uint(x);
  u += 0x7FFFu + ((u >> 16) & 1u);          // round-to-nearest-even
  return (unsigned short)(u >> 16);
}
static __device__ __forceinline__ float sigmoidf_(float x) { return 1.0f / (1.0f + __expf(-x)); }
static __device__ __forceinline__ float tanhf_(float x) {
  float t = __expf(-2.0f * x);
  return (1.0f - t) / (1.0f + t);
}

#define WMMA_BF16(A_, B_, C_) \
  __builtin_amdgcn_wmma_f32_16x16x32_bf16(false, (A_), false, (B_), (short)0, (C_), false, false)

// Stage nIter*4KB of contiguous global data into LDS via the CDNA5 async-to-LDS path.
// 256 threads x 16 B per issue. Tracked with ASYNCcnt; barrier before use.
static __device__ __forceinline__ void async_stage(const unsigned short* __restrict__ gbase,
                                                   unsigned short* sdst, int tid, int nIter) {
  const unsigned ldsBase = (unsigned)(unsigned long long)sdst;
#pragma unroll
  for (int i = 0; i < 16; ++i) {
    if (i >= nIter) break;
    const unsigned byteOff = (unsigned)(i * 4096 + tid * 16);
    asm volatile("global_load_async_to_lds_b128 %0, %1, %2"
                 :: "v"(ldsBase + byteOff), "v"(byteOff), "s"(gbase)
                 : "memory");
  }
  asm volatile("s_wait_asynccnt 0x0" ::: "memory");
  __syncthreads();
}

// Wave-level 32(M) x 64(N) x 1024(K) bf16 GEMM core, A from LDS, B from global.
// Ping-pong double buffering: no cross-iteration register copies.
static __device__ __forceinline__ void wave_gemm_32x64(const unsigned short* sA,  // LDS [32][1024]
                                                       const unsigned short* B0,  // [64 rows][1024]
                                                       int lane, v8f acc[2][4]) {
  v16bf a0[2], b0[4], a1[2], b1[4];
#pragma unroll
  for (int m = 0; m < 2; ++m) a0[m] = load_frag(sA + m * 16 * E, E, 0, lane);
#pragma unroll
  for (int t = 0; t < 4; ++t) b0[t] = load_frag(B0 + (size_t)t * 16 * E, E, 0, lane);

  for (int k0 = 0; k0 < E; k0 += 64) {
    // preload k0+32 into buffer 1 (always in range: k0 <= 960)
#pragma unroll
    for (int m = 0; m < 2; ++m) a1[m] = load_frag(sA + m * 16 * E, E, k0 + 32, lane);
#pragma unroll
    for (int t = 0; t < 4; ++t) b1[t] = load_frag(B0 + (size_t)t * 16 * E, E, k0 + 32, lane);
#pragma unroll
    for (int m = 0; m < 2; ++m)
#pragma unroll
      for (int t = 0; t < 4; ++t) acc[m][t] = WMMA_BF16(a0[m], b0[t], acc[m][t]);
    if (k0 + 64 < E) {                     // preload k0+64 into buffer 0
#pragma unroll
      for (int m = 0; m < 2; ++m) a0[m] = load_frag(sA + m * 16 * E, E, k0 + 64, lane);
#pragma unroll
      for (int t = 0; t < 4; ++t) b0[t] = load_frag(B0 + (size_t)t * 16 * E, E, k0 + 64, lane);
    }
#pragma unroll
    for (int m = 0; m < 2; ++m)
#pragma unroll
      for (int t = 0; t < 4; ++t) acc[m][t] = WMMA_BF16(a1[m], b1[t], acc[m][t]);
  }
}

// ---------------- prep kernels ----------------

__global__ void cast_f32_to_bf16(const float* __restrict__ src, unsigned short* __restrict__ dst, int n) {
  int i = blockIdx.x * blockDim.x + threadIdx.x;
  int stride = gridDim.x * blockDim.x;
  for (; i < n; i += stride) dst[i] = f2bf(src[i]);
}

__global__ void cast_emb_pad(const float* __restrict__ emb, unsigned short* __restrict__ dst) {
  long long i = (long long)blockIdx.x * blockDim.x + threadIdx.x;
  const long long n = (long long)NPAD * E;
  const long long stride = (long long)gridDim.x * blockDim.x;
  for (; i < n; i += stride) {
    long long row = i >> 10;  // /E
    dst[i] = (row < VOCAB) ? f2bf(emb[(row << 10) + (i & (E - 1))]) : (unsigned short)0;
  }
}

__global__ void embed_gather(const int* __restrict__ x, const float* __restrict__ emb,
                             unsigned short* __restrict__ xe) {
  const int row = blockIdx.x;            // b*T + t
  const int tok = x[row];
  const float* s = emb + (size_t)tok * E;
  unsigned short* d = xe + (size_t)row * E;
  for (int j = threadIdx.x; j < E; j += blockDim.x) d[j] = f2bf(s[j]);
}

__global__ void init_state(float* __restrict__ c, unsigned short* __restrict__ h0) {
  int i = blockIdx.x * blockDim.x + threadIdx.x;
  if (i < Bsz * H) { c[i] = 0.0f; h0[i] = 0; }
}

// ---------------- input-side GEMM: gates_x = xe @ w_ih^T + (b_ih + b_hh) ----------------
// grid (8, 128), block 256 (8 waves), 64 KB dynamic LDS (shared 32-row A tile).
// wave -> 32(M) x 64(N) C tile. launch_bounds(.,1): allow full VGPR budget, no spills.

__global__ void __launch_bounds__(256, 1)
gemm_gates(const unsigned short* __restrict__ A,   // xe  [4096,1024] bf16
           const unsigned short* __restrict__ Bw,  // w_ih[4096,1024] bf16
           const float* __restrict__ b_ih, const float* __restrict__ b_hh,
           float* __restrict__ C) {                // [4096,4096] f32
  extern __shared__ unsigned short sA[];           // [32][1024]
  const int tid  = threadIdx.x;
  const int lane = tid & 31;
  const int wave = tid >> 5;
  const int m0 = blockIdx.y * 32;
  async_stage(A + (size_t)m0 * E, sA, tid, 16);

  const int n0 = (blockIdx.x * 8 + wave) * 64;
  v8f acc[2][4] = {};
  wave_gemm_32x64(sA, Bw + (size_t)n0 * E, lane, acc);

  const int mrow = (lane >> 4) << 3;
  const int ncol = lane & 15;
#pragma unroll
  for (int m = 0; m < 2; ++m)
#pragma unroll
    for (int t = 0; t < 4; ++t) {
      const int n = n0 + t * 16 + ncol;
      const float bias = b_ih[n] + b_hh[n];
#pragma unroll
      for (int j = 0; j < 8; ++j)
        C[(size_t)(m0 + m * 16 + mrow + j) * G4H + n] = acc[m][t][j] + bias;
    }
}

// ---------------- one LSTM time step (launched 256x, serially) ----------------
// grid 64 x block 128 (4 waves). Block owns hidden tile n0 = blockIdx.x*16.
// K=1024 split across the 4 waves (256 each -> 8 k-steps, ping-pong pipelined),
// partials reduced through LDS, cell update fused on wave 0.

__global__ void __launch_bounds__(128, 1)
lstm_step(const unsigned short* __restrict__ h_prev, // [16,1024] bf16
          const unsigned short* __restrict__ Whh,    // [4096,1024] bf16
          const float* __restrict__ gatesx,          // [4096,4096] f32
          float* __restrict__ c,                     // [16,1024] f32
          unsigned short* __restrict__ h_next,       // [16,1024] bf16
          unsigned short* __restrict__ hs_bf,        // [4096,1024] bf16
          float* __restrict__ out_tail,              // d_out + LOGP_N
          int t) {
  __shared__ float red[4 * 4 * 8 * 32];    // [wave][gate][j][lane] = 16 KB
  const int lane  = threadIdx.x & 31;
  const int wave  = threadIdx.x >> 5;      // K-slice 0..3
  const int n0    = blockIdx.x * 16;
  const int kbase = wave * 256;

  v8f acc[4] = {};
  v16bf a0, b0[4], a1, b1[4];
  a0 = load_frag(h_prev, H, kbase, lane);
#pragma unroll
  for (int g = 0; g < 4; ++g)
    b0[g] = load_frag(Whh + (size_t)(g * H + n0) * H, H, kbase, lane);

#pragma unroll
  for (int i = 0; i < 4; ++i) {
    const int kA = kbase + i * 64;
    a1 = load_frag(h_prev, H, kA + 32, lane);
#pragma unroll
    for (int g = 0; g < 4; ++g)
      b1[g] = load_frag(Whh + (size_t)(g * H + n0) * H, H, kA + 32, lane);
#pragma unroll
    for (int g = 0; g < 4; ++g) acc[g] = WMMA_BF16(a0, b0[g], acc[g]);
    if (i < 3) {
      a0 = load_frag(h_prev, H, kA + 64, lane);
#pragma unroll
      for (int g = 0; g < 4; ++g)
        b0[g] = load_frag(Whh + (size_t)(g * H + n0) * H, H, kA + 64, lane);
    }
#pragma unroll
    for (int g = 0; g < 4; ++g) acc[g] = WMMA_BF16(a1, b1[g], acc[g]);
  }

#pragma unroll
  for (int g = 0; g < 4; ++g)
#pragma unroll
    for (int j = 0; j < 8; ++j)
      red[((wave * 4 + g) * 8 + j) * 32 + lane] = acc[g][j];
  __syncthreads();
  if (wave != 0) return;

  const int mrowBase = (lane >> 4) << 3;
  const int ncol = lane & 15;
#pragma unroll
  for (int j = 0; j < 8; ++j) {
    const int mb = mrowBase + j;             // batch index
    const int n  = n0 + ncol;                // hidden index
    float gsum[4];
#pragma unroll
    for (int g = 0; g < 4; ++g) {
      float s = 0.f;
#pragma unroll
      for (int w = 0; w < 4; ++w) s += red[((w * 4 + g) * 8 + j) * 32 + lane];
      gsum[g] = s;
    }
    const size_t grow = ((size_t)mb * Tlen + t) * (size_t)G4H;
    float gi = gsum[0] + gatesx[grow + 0 * H + n];
    float gf = gsum[1] + gatesx[grow + 1 * H + n];
    float gg = gsum[2] + gatesx[grow + 2 * H + n];
    float go = gsum[3] + gatesx[grow + 3 * H + n];
    float iv = sigmoidf_(gi), fv = sigmoidf_(gf), gv = tanhf_(gg), ov = sigmoidf_(go);
    float cn = fv * c[mb * H + n] + iv * gv;
    float hn = ov * tanhf_(cn);
    c[mb * H + n] = cn;
    unsigned short hb = f2bf(hn);
    h_next[mb * H + n] = hb;
    hs_bf[((size_t)mb * Tlen + t) * H + n] = hb;
    if (t == Tlen - 1) {
      out_tail[mb * H + n] = hn;             // hT [1,16,1024]
      out_tail[Bsz * H + mb * H + n] = cn;   // cT [1,16,1024]
    }
  }
}

// ---------------- decoder GEMM: logits = hs @ emb^T + b_dec (tied weights) ----------------
// grid (99, 128), block 256 (8 waves), 64 KB dynamic LDS (shared 32-row A tile).
// wave -> 32x64 C tile; N padded to 50304, stores guarded.

__global__ void __launch_bounds__(256, 1)
gemm_decoder(const unsigned short* __restrict__ A,   // hs_bf [4096,1024]
             const unsigned short* __restrict__ Bw,  // emb_pad [50304,1024]
             const float* __restrict__ b_dec,
             float* __restrict__ out) {              // d_out [4096,50257]
  extern __shared__ unsigned short sA[];             // [32][1024]
  const int tid  = threadIdx.x;
  const int lane = tid & 31;
  const int wave = tid >> 5;
  const int m0 = blockIdx.y * 32;
  async_stage(A + (size_t)m0 * E, sA, tid, 16);     // barrier inside, before any exit

  const int ngrp = blockIdx.x * 8 + wave;
  if (ngrp >= NGRP) return;                         // whole-wave uniform exit, after barrier
  const int n0 = ngrp * 64;
  v8f acc[2][4] = {};
  wave_gemm_32x64(sA, Bw + (size_t)n0 * E, lane, acc);

  const int mrow = (lane >> 4) << 3;
  const int ncol = lane & 15;
#pragma unroll
  for (int m = 0; m < 2; ++m)
#pragma unroll
    for (int t = 0; t < 4; ++t) {
      const int n = n0 + t * 16 + ncol;
      if (n < VOCAB) {
        const float bias = b_dec[n];
#pragma unroll
        for (int j = 0; j < 8; ++j)
          out[(size_t)(m0 + m * 16 + mrow + j) * VOCAB + n] = acc[m][t][j] + bias;
      }
    }
}

// ---------------- in-place row log-softmax over vocab ----------------

__global__ void log_softmax_rows(float* __restrict__ out) {
  __shared__ float red[256];
  float* p = out + (size_t)blockIdx.x * VOCAB;
  float m = -3.4e38f;
  for (int i = threadIdx.x; i < VOCAB; i += blockDim.x) m = fmaxf(m, p[i]);
  red[threadIdx.x] = m; __syncthreads();
  for (int s = 128; s > 0; s >>= 1) {
    if (threadIdx.x < s) red[threadIdx.x] = fmaxf(red[threadIdx.x], red[threadIdx.x + s]);
    __syncthreads();
  }
  m = red[0]; __syncthreads();
  float sum = 0.f;
  for (int i = threadIdx.x; i < VOCAB; i += blockDim.x) sum += __expf(p[i] - m);
  red[threadIdx.x] = sum; __syncthreads();
  for (int s = 128; s > 0; s >>= 1) {
    if (threadIdx.x < s) red[threadIdx.x] += red[threadIdx.x + s];
    __syncthreads();
  }
  const float lse = m + __logf(red[0]);
  for (int i = threadIdx.x; i < VOCAB; i += blockDim.x) p[i] = p[i] - lse;
}

// ---------------- launch ----------------

extern "C" void kernel_launch(void* const* d_in, const int* in_sizes, int n_in,
                              void* d_out, int out_size, void* d_ws, size_t ws_size,
                              hipStream_t stream) {
  (void)in_sizes; (void)n_in; (void)out_size; (void)ws_size;
  const int*   x     = (const int*)d_in[0];
  const float* emb   = (const float*)d_in[1];
  const float* w_ih  = (const float*)d_in[2];
  const float* w_hh  = (const float*)d_in[3];
  const float* b_ih  = (const float*)d_in[4];
  const float* b_hh  = (const float*)d_in[5];
  const float* b_dec = (const float*)d_in[6];
  float* out = (float*)d_out;

  char* ws = (char*)d_ws;
  size_t off = 0;
  auto alloc = [&](size_t bytes) -> void* {
    void* p = ws + off;
    off = (off + bytes + 255) & ~(size_t)255;
    return p;
  };
  unsigned short* wih_bf = (unsigned short*)alloc((size_t)G4H * E * 2);    // 8 MB
  unsigned short* whh_bf = (unsigned short*)alloc((size_t)G4H * H * 2);    // 8 MB
  unsigned short* emb_bf = (unsigned short*)alloc((size_t)NPAD * E * 2);   // 103 MB
  unsigned short* xe_bf  = (unsigned short*)alloc((size_t)BT * E * 2);     // 8 MB
  float*          gatesx = (float*)alloc((size_t)BT * G4H * 4);            // 64 MB
  unsigned short* hs_bf  = (unsigned short*)alloc((size_t)BT * H * 2);     // 8 MB
  float*          cbuf   = (float*)alloc((size_t)Bsz * H * 4);
  unsigned short* hbuf0  = (unsigned short*)alloc((size_t)Bsz * H * 2);
  unsigned short* hbuf1  = (unsigned short*)alloc((size_t)Bsz * H * 2);

  cast_f32_to_bf16<<<2048, 256, 0, stream>>>(w_ih, wih_bf, G4H * E);
  cast_f32_to_bf16<<<2048, 256, 0, stream>>>(w_hh, whh_bf, G4H * H);
  cast_emb_pad<<<8192, 256, 0, stream>>>(emb, emb_bf);
  embed_gather<<<BT, 256, 0, stream>>>(x, emb, xe_bf);
  init_state<<<(Bsz * H + 255) / 256, 256, 0, stream>>>(cbuf, hbuf0);

  gemm_gates<<<dim3(8, BT / 32), 256, 65536, stream>>>(xe_bf, wih_bf, b_ih, b_hh, gatesx);

  float* out_tail = out + LOGP_N;
  for (int t = 0; t < Tlen; ++t) {
    const unsigned short* hp = (t & 1) ? hbuf1 : hbuf0;
    unsigned short*       hn = (t & 1) ? hbuf0 : hbuf1;
    lstm_step<<<dim3(64), 128, 0, stream>>>(hp, whh_bf, gatesx, cbuf, hn, hs_bf, out_tail, t);
  }

  gemm_decoder<<<dim3(99, BT / 32), 256, 65536, stream>>>(hs_bf, emb_bf, b_dec, out);
  log_softmax_rows<<<BT, 256, 0, stream>>>(out);
}
